// GraphEncoder_82575041233259
// MI455X (gfx1250) — matrix operand order repaired and checked by reading, not compiled, not run on hardware
//
#include <hip/hip_runtime.h>
#include <hip/hip_bf16.h>
#include <math.h>

// ---------------- constants ----------------
#define B_GRAPHS 128
#define NPG0     512
#define NMAX     (B_GRAPHS * NPG0)   // 65536
#define E_EDGES  262144
#define IN_FEATS 128
#define EDGE_FEATS 32
#define EMB      64
#define HEADS    4
#define HC       256                 // EMB*HEADS
#define NEG_BIG  (-3.0e38f)

typedef __attribute__((ext_vector_type(16))) __bf16   v16bf;
typedef __attribute__((ext_vector_type(8)))  float    v8f;
typedef __attribute__((ext_vector_type(4)))  unsigned v4u;

static __device__ __forceinline__ unsigned ord_f32(float f) {
  unsigned u = __float_as_uint(f);
  return (u & 0x80000000u) ? ~u : (u | 0x80000000u);
}
static __device__ __forceinline__ float unord_f32(unsigned u) {
  return (u & 0x80000000u) ? __uint_as_float(u ^ 0x80000000u)
                           : __uint_as_float(~u);
}

// ---------------- utility kernels ----------------
__global__ void ge_fill_f32(float* p, float v, int n) {
  int i = blockIdx.x * blockDim.x + threadIdx.x;
  if (i < n) p[i] = v;
}
__global__ void ge_fill_u32(unsigned* p, unsigned v, int n) {
  int i = blockIdx.x * blockDim.x + threadIdx.x;
  if (i < n) p[i] = v;
}
__global__ void ge_f32_to_bf16(const float* __restrict__ s, __bf16* __restrict__ d, int n) {
  int i = blockIdx.x * blockDim.x + threadIdx.x;
  if (i < n) d[i] = (__bf16)s[i];
}
__global__ void ge_init_edges(const int* __restrict__ ei, int* src, int* dst, int* emk, int e) {
  int i = blockIdx.x * blockDim.x + threadIdx.x;
  if (i >= e) return;
  src[i] = ei[i];
  dst[i] = ei[e + i];
  emk[i] = 1;
}

// ---- pack fp32 weight W[K,Ncol] into bf16 B-fragment layout ----
// P layout: [ktile][ntile][lane 0..31][16 contiguous bf16]
// lane half (lane>>4) selects K kb..kb+15 vs kb+16..kb+31; lane&15 = column.
__global__ void ge_pack_w(const float* __restrict__ W, __bf16* __restrict__ P,
                          int K, int Ncol) {
  int idx = blockIdx.x * blockDim.x + threadIdx.x;
  int total = K * Ncol;
  if (idx >= total) return;
  int e     = idx & 15;
  int lane  = (idx >> 4) & 31;
  int ntile = (idx >> 9) % (Ncol >> 4);
  int ktile = idx / (512 * (Ncol >> 4));
  int half  = lane >> 4;
  int col   = ntile * 16 + (lane & 15);
  int k     = ktile * 32 + half * 16 + e;
  P[idx] = (__bf16)W[(size_t)k * Ncol + col];
}

// ---------------- WMMA GEMM: C[n,Ncol] = A[n,K](bf16) @ W(packed bf16) + bias, opt ReLU ---
// One wave computes a 16x64 output tile: 4 accumulators reuse one A fragment.
// A fragment (16x32 bf16, ISA 7.12.2): each lane = two 16B contiguous row chunks.
// B fragments: 32 contiguous bytes per lane from the pre-packed layout; all four
// fragments are loaded before the WMMAs so the waits can be partial (overlap).
__global__ void ge_wmma_gemm(const __bf16* __restrict__ A,
                             const __bf16* __restrict__ Wp,
                             const float*  __restrict__ bias,
                             float* __restrict__ C,
                             int n, int K, int Ncol, int do_relu)
{
  const int lane   = threadIdx.x;        // 0..31
  const int mt     = blockIdx.y * 16;
  const int nt0    = blockIdx.x * 4;     // first 16-col tile of this wave's 64-col strip
  const int half   = lane >> 4;
  const int l15    = lane & 15;
  const int row    = mt + l15;
  const int ntiles = Ncol >> 4;

  v8f accs[4] = {};
  const __bf16* arow = A + (size_t)row * K;

  for (int kb = 0; kb < K; kb += 32) {
    union { v4u u[2]; v16bf v; } af;
    union { v4u u[2]; v16bf v; } bfr[4];
    const __bf16* bp = Wp + ((size_t)(kb >> 5) * ntiles + nt0) * 512 + lane * 16;
    // issue all loads first (A + 4 B fragments), WMMAs after -> partial loadcnt waits
    af.u[0] = *(const v4u*)(arow + kb + half * 16);
    af.u[1] = *(const v4u*)(arow + kb + half * 16 + 8);
#pragma unroll
    for (int t = 0; t < 4; ++t) {
      bfr[t].u[0] = *(const v4u*)(bp + t * 512);
      bfr[t].u[1] = *(const v4u*)(bp + t * 512 + 8);
    }
    __builtin_prefetch(arow + kb + 32, 0, 1);                  // next A chunk
    __builtin_prefetch(bp + (size_t)ntiles * 512, 0, 1);       // next packed-B strip
#pragma unroll
    for (int t = 0; t < 4; ++t) {
      accs[t] = __builtin_amdgcn_wmma_f32_16x16x32_bf16(false, af.v, false, bfr[t].v,
                                                        (short)0, accs[t], false, false);
    }
  }

  const int colbase = nt0 * 16 + l15;
#pragma unroll
  for (int t = 0; t < 4; ++t) {
    int col = colbase + t * 16;
    float bv = bias ? bias[col] : 0.0f;
#pragma unroll
    for (int j = 0; j < 8; ++j) {
      int r = mt + j + 8 * half;           // C: VGPR j => M=j (lanes0-15) / M=j+8
      float val = accs[t][j] + bv;
      if (do_relu) val = fmaxf(val, 0.0f);
      C[(size_t)r * Ncol + col] = val;
    }
  }
}

// ---------------- attention kernels ----------------
__global__ void ge_attn_alpha(const float* __restrict__ q, const float* __restrict__ k,
                              const float* __restrict__ e,
                              const int* __restrict__ src, const int* __restrict__ dst,
                              const int* __restrict__ emk,
                              float* __restrict__ alpha, unsigned* __restrict__ mmax,
                              int total)
{
  int t = blockIdx.x * blockDim.x + threadIdx.x;
  if (t >= total) return;
  int ed = t >> 2, h = t & 3;
  if (!emk[ed]) { alpha[t] = NEG_BIG; return; }
  int s = src[ed], d = dst[ed];
  const float* qp = q + (size_t)d * HC + h * EMB;
  const float* kp = k + (size_t)s * HC + h * EMB;
  const float* ep = e + (size_t)ed * HC + h * EMB;
  float acc = 0.f;
  for (int c = 0; c < EMB; ++c) acc += qp[c] * (kp[c] + ep[c]);
  float a = acc * 0.125f;                  // 1/sqrt(EMB)
  alpha[t] = a;
  atomicMax(&mmax[d * HEADS + h], ord_f32(a));
}

__global__ void ge_finalize_m(unsigned* m, int total) {
  int i = blockIdx.x * blockDim.x + threadIdx.x;
  if (i >= total) return;
  float f = unord_f32(m[i]);
  if (!(f > NEG_BIG) || !(f < -NEG_BIG)) f = 0.0f;   // non-finite -> 0
  m[i] = __float_as_uint(f);
}

__global__ void ge_attn_p(float* __restrict__ alpha, const unsigned* __restrict__ m,
                          const int* __restrict__ dst, const int* __restrict__ emk,
                          float* __restrict__ den, int total)
{
  int t = blockIdx.x * blockDim.x + threadIdx.x;
  if (t >= total) return;
  int ed = t >> 2, h = t & 3;
  if (!emk[ed]) { alpha[t] = 0.f; return; }
  int d = dst[ed];
  float p = __expf(alpha[t] - __uint_as_float(m[d * HEADS + h]));
  alpha[t] = p;
  atomicAdd(&den[d * HEADS + h], p);
}

__global__ void ge_attn_norm(float* __restrict__ alpha, const float* __restrict__ den,
                             const int* __restrict__ dst, const int* __restrict__ emk,
                             int total)
{
  int t = blockIdx.x * blockDim.x + threadIdx.x;
  if (t >= total) return;
  int ed = t >> 2, h = t & 3;
  if (!emk[ed]) return;
  alpha[t] /= fmaxf(den[dst[ed] * HEADS + h], 1e-16f);
}

__global__ void ge_aggregate(const float* __restrict__ p4, const float* __restrict__ v,
                             const float* __restrict__ e,
                             const int* __restrict__ src, const int* __restrict__ dst,
                             const int* __restrict__ emk, float* __restrict__ out,
                             int total)
{
  int t = blockIdx.x * blockDim.x + threadIdx.x;
  if (t >= total) return;
  int ed = t >> 6;
  int c4 = (t & 63) << 2;                  // 4 channels per thread
  if (!emk[ed]) return;
  float p = p4[ed * HEADS + (c4 >> 6)];
  if (p == 0.f) return;
  int s = src[ed], d = dst[ed];
  const float* vp = v + (size_t)s * HC + c4;
  const float* ep = e + (size_t)ed * HC + c4;
  float* op = out + (size_t)d * HC + c4;
#pragma unroll
  for (int i = 0; i < 4; ++i) atomicAdd(&op[i], (vp[i] + ep[i]) * p);
}

__global__ void ge_beta_combine(const float* __restrict__ out, const float* __restrict__ xr,
                                const float* __restrict__ Wb, float* __restrict__ h, int n)
{
  int i = blockIdx.x * blockDim.x + threadIdx.x;
  if (i >= n) return;
  const float* op = out + (size_t)i * HC;
  const float* xp = xr  + (size_t)i * HC;
  float acc = 0.f;
  for (int c = 0; c < HC; ++c) {
    float o = op[c], x = xp[c];
    acc += o * Wb[c] + x * Wb[HC + c] + (o - x) * Wb[2 * HC + c];
  }
  float beta = 1.0f / (1.0f + __expf(-acc));
  for (int c = 0; c < HC; ++c)
    h[(size_t)i * HC + c] = beta * xp[c] + (1.f - beta) * op[c];
}

// ---------------- batchnorm ----------------
__global__ void ge_bn_stats(const float* __restrict__ x, float* mu, float* var, int n) {
  __shared__ float ss[256], sq[256];
  int c = blockIdx.x, tid = threadIdx.x;
  float s = 0.f, q = 0.f;
  for (int i = tid; i < n; i += 256) {
    float v = x[(size_t)i * EMB + c];
    s += v; q += v * v;
  }
  ss[tid] = s; sq[tid] = q;
  __syncthreads();
  for (int st = 128; st > 0; st >>= 1) {
    if (tid < st) { ss[tid] += ss[tid + st]; sq[tid] += sq[tid + st]; }
    __syncthreads();
  }
  if (tid == 0) {
    float m = ss[0] / (float)n;
    mu[c] = m;
    var[c] = sq[0] / (float)n - m * m;
  }
}
__global__ void ge_bn_apply(float* x, const float* mu, const float* var,
                            const float* g, const float* b, int total) {
  int i = blockIdx.x * blockDim.x + threadIdx.x;
  if (i >= total) return;
  int c = i & (EMB - 1);
  x[i] = (x[i] - mu[c]) * rsqrtf(var[c] + 1e-5f) * g[c] + b[c];
}

// ---------------- topk pooling ----------------
__global__ void ge_topk(const float* __restrict__ x, const float* __restrict__ w,
                        float* __restrict__ xnew, int* __restrict__ nidx,
                        int npg, int kk)
{
  __shared__ float ssc[512];
  __shared__ int   sid[512];
  __shared__ float swn;
  int g = blockIdx.x, tid = threadIdx.x;
  if (tid == 0) {
    float s = 0.f;
    for (int i = 0; i < EMB; ++i) s += w[i] * w[i];
    swn = sqrtf(s);
  }
  __syncthreads();
  float sc = NEG_BIG;
  if (tid < npg) {
    const float* xp = x + (size_t)(g * npg + tid) * EMB;
    float acc = 0.f;
    for (int i = 0; i < EMB; ++i) acc += xp[i] * w[i];
    sc = tanhf(acc / swn);
    nidx[g * npg + tid] = -1;
  }
  ssc[tid] = sc; sid[tid] = tid;
  __syncthreads();
  for (int k2 = 2; k2 <= 512; k2 <<= 1) {
    for (int j = k2 >> 1; j > 0; j >>= 1) {
      int ixj = tid ^ j;
      if (ixj > tid) {
        bool desc = ((tid & k2) == 0);
        float a = ssc[tid], b = ssc[ixj];
        if (desc ? (a < b) : (a > b)) {
          ssc[tid] = b; ssc[ixj] = a;
          int tt = sid[tid]; sid[tid] = sid[ixj]; sid[ixj] = tt;
        }
      }
      __syncthreads();
    }
  }
  if (tid < kk) {
    int oldn = g * npg + sid[tid];
    int newn = g * kk + tid;
    float sv = ssc[tid];
    nidx[oldn] = newn;
    const float* xp = x + (size_t)oldn * EMB;
    float* yp = xnew + (size_t)newn * EMB;
    for (int i = 0; i < EMB; ++i) yp[i] = xp[i] * sv;
  }
}

__global__ void ge_remap_edges(const int* __restrict__ nidx, int* src, int* dst,
                               int* emk, int e) {
  int i = blockIdx.x * blockDim.x + threadIdx.x;
  if (i >= e) return;
  if (emk[i]) {
    int sn = nidx[src[i]], dn = nidx[dst[i]];
    if (sn >= 0 && dn >= 0) { src[i] = sn; dst[i] = dn; }
    else { src[i] = 0; dst[i] = 0; emk[i] = 0; }
  } else { src[i] = 0; dst[i] = 0; }
}

__global__ void ge_pool_add(const float* __restrict__ x, float* __restrict__ out, int npg) {
  int g = blockIdx.x, c = threadIdx.x;   // c < EMB
  const float* xp = x + (size_t)g * npg * EMB + c;
  float mx = NEG_BIG, sm = 0.f;
  for (int i = 0; i < npg; ++i) {
    float v = xp[(size_t)i * EMB];
    mx = fmaxf(mx, v); sm += v;
  }
  out[g * 2 * EMB + c]       += mx;
  out[g * 2 * EMB + EMB + c] += sm / (float)npg;
}

// ---------------- host orchestration ----------------
struct GeScratch {
  __bf16 *abf, *eabf;
  float *q, *k, *v, *sk, *outb, *hb, *ebuf, *alpha, *den;
  unsigned* m;
  int *src, *dst, *emk;
};

static inline int gcdiv(int a, int b) { return (a + b - 1) / b; }

static void ge_tconv(hipStream_t st, const GeScratch& w,
                     const float* x, int n, int in_feats,
                     const __bf16* Wqb, const __bf16* Wkb, const __bf16* Wvb,
                     const __bf16* Wskb, const __bf16* Web,
                     const float* bq, const float* bk, const float* bv,
                     const float* bsk, const float* Wbeta)
{
  ge_f32_to_bf16<<<gcdiv(n * in_feats, 256), 256, 0, st>>>(x, w.abf, n * in_feats);
  dim3 blk(32);
  dim3 gq(HC / 64, n / 16);     // each wave: 16 rows x 64 cols
  ge_wmma_gemm<<<gq, blk, 0, st>>>(w.abf, Wqb,  bq,  w.q,  n, in_feats, HC, 0);
  ge_wmma_gemm<<<gq, blk, 0, st>>>(w.abf, Wkb,  bk,  w.k,  n, in_feats, HC, 0);
  ge_wmma_gemm<<<gq, blk, 0, st>>>(w.abf, Wvb,  bv,  w.v,  n, in_feats, HC, 0);
  ge_wmma_gemm<<<gq, blk, 0, st>>>(w.abf, Wskb, bsk, w.sk, n, in_feats, HC, 0);
  dim3 ge(HC / 64, E_EDGES / 16);
  ge_wmma_gemm<<<ge, blk, 0, st>>>(w.eabf, Web, (const float*)nullptr, w.ebuf,
                                   E_EDGES, EDGE_FEATS, HC, 0);
  ge_fill_u32<<<gcdiv(n * HEADS, 256), 256, 0, st>>>(w.m, 0x007FFFFFu, n * HEADS); // ord(-inf)
  ge_fill_f32<<<gcdiv(n * HEADS, 256), 256, 0, st>>>(w.den, 0.f, n * HEADS);
  int ta = E_EDGES * HEADS;
  ge_attn_alpha<<<gcdiv(ta, 256), 256, 0, st>>>(w.q, w.k, w.ebuf, w.src, w.dst, w.emk,
                                                w.alpha, w.m, ta);
  ge_finalize_m<<<gcdiv(n * HEADS, 256), 256, 0, st>>>(w.m, n * HEADS);
  ge_attn_p<<<gcdiv(ta, 256), 256, 0, st>>>(w.alpha, w.m, w.dst, w.emk, w.den, ta);
  ge_attn_norm<<<gcdiv(ta, 256), 256, 0, st>>>(w.alpha, w.den, w.dst, w.emk, ta);
  ge_fill_f32<<<gcdiv(n * HC, 256), 256, 0, st>>>(w.outb, 0.f, n * HC);
  int tg = E_EDGES * 64;
  ge_aggregate<<<gcdiv(tg, 256), 256, 0, st>>>(w.alpha, w.v, w.ebuf, w.src, w.dst, w.emk,
                                               w.outb, tg);
  ge_beta_combine<<<gcdiv(n, 256), 256, 0, st>>>(w.outb, w.sk, Wbeta, w.hb, n);
}

static void ge_post(hipStream_t st, const GeScratch& w, int n,
                    const __bf16* Wtb, const float* bt,
                    const float* g, const float* b,
                    float* xout, float* mu, float* var)
{
  ge_f32_to_bf16<<<gcdiv(n * HC, 256), 256, 0, st>>>(w.hb, w.abf, n * HC);
  dim3 gt(EMB / 64, n / 16);    // EMB=64 -> one 64-col strip
  ge_wmma_gemm<<<gt, dim3(32), 0, st>>>(w.abf, Wtb, bt, xout, n, HC, EMB, 1); // + ReLU
  ge_bn_stats<<<EMB, 256, 0, st>>>(xout, mu, var, n);
  ge_bn_apply<<<gcdiv(n * EMB, 256), 256, 0, st>>>(xout, mu, var, g, b, n * EMB);
}

extern "C" void kernel_launch(void* const* d_in, const int* in_sizes, int n_in,
                              void* d_out, int out_size, void* d_ws, size_t ws_size,
                              hipStream_t stream)
{
  (void)in_sizes; (void)n_in; (void)ws_size;
  // ---- inputs (setup_inputs order) ----
  const float* x_in    = (const float*)d_in[0];
  const float* eattr   = (const float*)d_in[1];
  const int*   eindex  = (const int*)d_in[2];
  // d_in[3] batch_index (unused: graphs contiguous)
  const float* Wq1 = (const float*)d_in[4];   const float* bq1 = (const float*)d_in[5];
  const float* Wk1 = (const float*)d_in[6];   const float* bk1 = (const float*)d_in[7];
  const float* Wv1 = (const float*)d_in[8];   const float* bv1 = (const float*)d_in[9];
  const float* We1 = (const float*)d_in[10];
  const float* Wsk1 = (const float*)d_in[11]; const float* bsk1 = (const float*)d_in[12];
  const float* Wbeta1 = (const float*)d_in[13];
  const float* Wt1 = (const float*)d_in[14];  const float* bt1 = (const float*)d_in[15];
  const float* bng1 = (const float*)d_in[16]; const float* bnb1 = (const float*)d_in[17];
  const float* Wq = (const float*)d_in[18];   const float* bq = (const float*)d_in[19];
  const float* Wk = (const float*)d_in[20];   const float* bk = (const float*)d_in[21];
  const float* Wv = (const float*)d_in[22];   const float* bv = (const float*)d_in[23];
  const float* We = (const float*)d_in[24];
  const float* Wsk = (const float*)d_in[25];  const float* bsk = (const float*)d_in[26];
  const float* Wbeta = (const float*)d_in[27];
  const float* Wt = (const float*)d_in[28];   const float* bt = (const float*)d_in[29];
  const float* bng = (const float*)d_in[30];  const float* bnb = (const float*)d_in[31];
  const float* pw0 = (const float*)d_in[32];  const float* pw2 = (const float*)d_in[33];

  // ---- workspace bump allocator ----
  char* base = (char*)d_ws;
  size_t off = 0;
  auto walloc = [&](size_t bytes) -> void* {
    void* p = base + off;
    off += (bytes + 255) & ~(size_t)255;
    return p;
  };

  __bf16* Wq1b  = (__bf16*)walloc((size_t)IN_FEATS * HC * 2);
  __bf16* Wk1b  = (__bf16*)walloc((size_t)IN_FEATS * HC * 2);
  __bf16* Wv1b  = (__bf16*)walloc((size_t)IN_FEATS * HC * 2);
  __bf16* Wsk1b = (__bf16*)walloc((size_t)IN_FEATS * HC * 2);
  __bf16* We1b  = (__bf16*)walloc((size_t)EDGE_FEATS * HC * 2);
  __bf16* Wt1b  = (__bf16*)walloc((size_t)HC * EMB * 2);
  __bf16* Wqb   = (__bf16*)walloc((size_t)3 * EMB * HC * 2);
  __bf16* Wkb   = (__bf16*)walloc((size_t)3 * EMB * HC * 2);
  __bf16* Wvb   = (__bf16*)walloc((size_t)3 * EMB * HC * 2);
  __bf16* Wskb  = (__bf16*)walloc((size_t)3 * EMB * HC * 2);
  __bf16* Web   = (__bf16*)walloc((size_t)3 * EDGE_FEATS * HC * 2);
  __bf16* Wtb   = (__bf16*)walloc((size_t)3 * HC * EMB * 2);

  GeScratch w;
  w.abf  = (__bf16*)walloc((size_t)NMAX * HC * 2);
  w.eabf = (__bf16*)walloc((size_t)E_EDGES * EDGE_FEATS * 2);
  w.q    = (float*)walloc((size_t)NMAX * HC * 4);
  w.k    = (float*)walloc((size_t)NMAX * HC * 4);
  w.v    = (float*)walloc((size_t)NMAX * HC * 4);
  w.sk   = (float*)walloc((size_t)NMAX * HC * 4);
  w.outb = (float*)walloc((size_t)NMAX * HC * 4);
  w.hb   = w.q;  // q is dead once alpha is formed; reuse for beta output
  w.ebuf = (float*)walloc((size_t)E_EDGES * HC * 4);
  w.alpha = (float*)walloc((size_t)E_EDGES * HEADS * 4);
  w.m    = (unsigned*)walloc((size_t)NMAX * HEADS * 4);
  w.den  = (float*)walloc((size_t)NMAX * HEADS * 4);
  w.src  = (int*)walloc((size_t)E_EDGES * 4);
  w.dst  = (int*)walloc((size_t)E_EDGES * 4);
  w.emk  = (int*)walloc((size_t)E_EDGES * 4);
  float* xa = (float*)walloc((size_t)NMAX * EMB * 4);
  float* xb = (float*)walloc((size_t)NMAX * EMB * 4);
  float* mu  = (float*)walloc(EMB * 4);
  float* var = (float*)walloc(EMB * 4);
  int* nidx  = (int*)walloc((size_t)NMAX * 4);

  auto packw = [&](const float* Wf, __bf16* P, int K, int Ncol) {
    ge_pack_w<<<gcdiv(K * Ncol, 256), 256, 0, stream>>>(Wf, P, K, Ncol);
  };

  // ---- weight packing: f32 -> bf16 fragment layout (deterministic every call) ----
  packw(Wq1, Wq1b, IN_FEATS, HC);   packw(Wk1, Wk1b, IN_FEATS, HC);
  packw(Wv1, Wv1b, IN_FEATS, HC);   packw(Wsk1, Wsk1b, IN_FEATS, HC);
  packw(We1, We1b, EDGE_FEATS, HC); packw(Wt1, Wt1b, HC, EMB);
  for (int i = 0; i < 3; ++i) {
    packw(Wq  + (size_t)i * EMB * HC,        Wqb  + (size_t)i * EMB * HC,        EMB, HC);
    packw(Wk  + (size_t)i * EMB * HC,        Wkb  + (size_t)i * EMB * HC,        EMB, HC);
    packw(Wv  + (size_t)i * EMB * HC,        Wvb  + (size_t)i * EMB * HC,        EMB, HC);
    packw(Wsk + (size_t)i * EMB * HC,        Wskb + (size_t)i * EMB * HC,        EMB, HC);
    packw(We  + (size_t)i * EDGE_FEATS * HC, Web  + (size_t)i * EDGE_FEATS * HC, EDGE_FEATS, HC);
    packw(Wt  + (size_t)i * HC * EMB,        Wtb  + (size_t)i * HC * EMB,        HC, EMB);
  }
  ge_f32_to_bf16<<<gcdiv(E_EDGES * EDGE_FEATS, 256), 256, 0, stream>>>(
      eattr, w.eabf, E_EDGES * EDGE_FEATS);

  ge_fill_f32<<<gcdiv(out_size, 256), 256, 0, stream>>>((float*)d_out, 0.f, out_size);
  ge_init_edges<<<gcdiv(E_EDGES, 256), 256, 0, stream>>>(eindex, w.src, w.dst, w.emk, E_EDGES);

  // ---- first TransformerConv (IN_FEATS=128) ----
  ge_tconv(stream, w, x_in, NMAX, IN_FEATS, Wq1b, Wk1b, Wv1b, Wsk1b, We1b,
           bq1, bk1, bv1, bsk1, Wbeta1);
  ge_post(stream, w, NMAX, Wt1b, bt1, bng1, bnb1, xa, mu, var);

  // ---- 3 layers with pooling at 0 and 2 ----
  int npg = NPG0, n = NMAX;
  for (int i = 0; i < 3; ++i) {
    ge_tconv(stream, w, xa, n, EMB,
             Wqb + (size_t)i * EMB * HC, Wkb + (size_t)i * EMB * HC,
             Wvb + (size_t)i * EMB * HC, Wskb + (size_t)i * EMB * HC,
             Web + (size_t)i * EDGE_FEATS * HC,
             bq + i * HC, bk + i * HC, bv + i * HC, bsk + i * HC,
             Wbeta + i * 3 * HC);
    ge_post(stream, w, n, Wtb + (size_t)i * HC * EMB, bt + i * EMB,
            bng + i * EMB, bnb + i * EMB, xa, mu, var);
    if (i == 0 || i == 2) {
      int kk = npg / 2;
      const float* pw = (i == 0) ? pw0 : pw2;
      ge_topk<<<B_GRAPHS, 512, 0, stream>>>(xa, pw, xb, nidx, npg, kk);
      ge_remap_edges<<<gcdiv(E_EDGES, 256), 256, 0, stream>>>(nidx, w.src, w.dst, w.emk, E_EDGES);
      npg = kk;
      n = B_GRAPHS * npg;
      float* t = xa; xa = xb; xb = t;
      ge_pool_add<<<B_GRAPHS, EMB, 0, stream>>>(xa, (float*)d_out, npg);
    }
  }
}
